// TorusEnglishDiffusion_75505525064338
// MI455X (gfx1250) — matrix operand order repaired and checked by reading, not compiled, not run on hardware
//
#include <hip/hip_runtime.h>
#include <hip/hip_bf16.h>

// Problem dims (fixed by reference)
#define B_   8
#define L_   1024
#define D_   1024
#define H_   16
#define HD_  64
#define DFF_ 4096
#define M_   (B_ * L_)   // 8192 rows
#define PAD_ 40          // LDS row pitch in bf16 elems (80B) -> bank-conflict-free b128 reads

typedef __bf16 bf16_t;
typedef bf16_t bf16x16 __attribute__((ext_vector_type(16)));
typedef float  f32x8   __attribute__((ext_vector_type(8)));
typedef int    v4i     __attribute__((ext_vector_type(4)));

#ifndef __has_builtin
#define __has_builtin(x) 0
#endif

#if __has_builtin(__builtin_amdgcn_global_load_async_to_lds_b128)
#define HAVE_ASYNC_LDS 1
typedef __attribute__((address_space(1))) v4i g_v4i;   // global (addrspace 1)
typedef __attribute__((address_space(3))) v4i l_v4i;   // LDS (addrspace 3)
#define ASYNC_CP16(g, l) \
  __builtin_amdgcn_global_load_async_to_lds_b128((g_v4i*)(g), (l_v4i*)(l), 0, 0)
#else
#define HAVE_ASYNC_LDS 0
#define ASYNC_CP16(g, l) \
  do { *reinterpret_cast<uint4*>(l) = *reinterpret_cast<const uint4*>(g); } while (0)
#endif

#if __has_builtin(__builtin_amdgcn_s_wait_asynccnt)
#define WAIT_ASYNC0() __builtin_amdgcn_s_wait_asynccnt(0)
#else
#define WAIT_ASYNC0() asm volatile("s_wait_asynccnt 0x0" ::: "memory")
#endif

union Frag {
  bf16x16 v;
  bf16_t  h[16];
  uint4   u[2];
};

// A operand (16x32, 16-bit): lane<16 holds K 0..15 contiguous, lane>=16 holds K 16..31
__device__ __forceinline__ Frag load_a_frag(const bf16_t* p, int lane) {
  Frag f;
  const bf16_t* q = p + (lane & 16);                  // +16 elems for upper half-wave
  f.u[0] = *reinterpret_cast<const uint4*>(q);        // K base .. base+7
  f.u[1] = *reinterpret_cast<const uint4*>(q + 8);    // K base+8 .. base+15
  return f;
}

// B operand (32x16, 16-bit): lane<16 holds K 0..15 contiguous, lane>=16 holds K 16..31
__device__ __forceinline__ Frag load_b_frag(const bf16_t* p, int lane) {
  Frag f;
  const bf16_t* q = p + (lane & 16);                  // +16 elems for upper half-wave
  f.u[0] = *reinterpret_cast<const uint4*>(q);
  f.u[1] = *reinterpret_cast<const uint4*>(q + 8);
  return f;
}

__device__ __forceinline__ f32x8 wmma_bf16(const Frag& a, const Frag& b, f32x8 c) {
  return __builtin_amdgcn_wmma_f32_16x16x32_bf16(false, a.v, false, b.v, (short)0, c, false, false);
}

__device__ __forceinline__ float gelu_exact(float x) {
  return 0.5f * x * (1.0f + erff(x * 0.70710678118654752f));
}

// ---------------------------------------------------------------------------
__global__ void f2bf_kernel(const float* __restrict__ in, bf16_t* __restrict__ out, int n) {
  int i = blockIdx.x * blockDim.x + threadIdx.x;
  if (i < n) out[i] = (bf16_t)in[i];
}

__global__ void gelu_vec_kernel(const float* __restrict__ in, float* __restrict__ out, int n) {
  int i = blockIdx.x * blockDim.x + threadIdx.x;
  if (i < n) out[i] = gelu_exact(in[i]);
}

// mask (bool bytes) -> additive float bias (0 or -3e38)
__global__ void maskbias_kernel(const unsigned char* __restrict__ m, float* __restrict__ mb, int n) {
  int i = blockIdx.x * blockDim.x + threadIdx.x;
  if (i < n) mb[i] = m[i] ? 0.0f : -3.0e38f;
}

// tp[b, j] = gelu_t[b,:] . w_time[j,:] + b_time[j]   (tiny: 33 MFLOP)
__global__ void time_proj_kernel(const float* __restrict__ gt, const float* __restrict__ w_time,
                                 const float* __restrict__ b_time, float* __restrict__ tp) {
  int j = blockIdx.x * blockDim.x + threadIdx.x;
  int b   = j >> 11;
  int col = j & (2 * D_ - 1);
  const float* wr = w_time + (size_t)col * D_;
  const float* gr = gt + (size_t)b * D_;
  float acc = b_time[col];
  for (int k = 0; k < D_; ++k) acc += gr[k] * wr[k];
  tp[(size_t)b * (2 * D_) + col] = acc;
}

// ---------------------------------------------------------------------------
// LayerNorm (+ optional adaLN modulation) -> bf16, one wave per row.
__global__ __launch_bounds__(256) void ln_mod_kernel(const float* __restrict__ x,
                                                     const float* __restrict__ g,
                                                     const float* __restrict__ bb,
                                                     const float* __restrict__ tp,
                                                     bf16_t* __restrict__ out,
                                                     int useMod) {
  int lane = threadIdx.x & 31;
  int wv   = threadIdx.x >> 5;
  int row  = blockIdx.x * 8 + wv;
  int b    = row >> 10;
  const float* xr = x + (size_t)row * D_;

  float xs[32];
  float mu = 0.f;
#pragma unroll
  for (int i = 0; i < 32; ++i) { xs[i] = xr[lane + i * 32]; mu += xs[i]; }
#pragma unroll
  for (int off = 16; off; off >>= 1) mu += __shfl_xor(mu, off);
  mu *= (1.0f / D_);
  float var = 0.f;
#pragma unroll
  for (int i = 0; i < 32; ++i) { float d = xs[i] - mu; var += d * d; }
#pragma unroll
  for (int off = 16; off; off >>= 1) var += __shfl_xor(var, off);
  var *= (1.0f / D_);
  float rs = rsqrtf(var + 1e-5f);

#pragma unroll
  for (int i = 0; i < 32; ++i) {
    int d = lane + i * 32;
    float v = (xs[i] - mu) * rs * g[d] + bb[d];
    if (useMod) {
      float sc = tp[(size_t)b * (2 * D_) + d];
      float sh = tp[(size_t)b * (2 * D_) + D_ + d];
      v = v * (1.0f + sc) + sh;
    }
    out[(size_t)row * D_ + d] = (bf16_t)v;
  }
}

// ---------------------------------------------------------------------------
// Tiled WMMA GEMM with double-buffered async-to-LDS staging.
// C[m,n] = sum_k A[m,k] * W[n,k]   (A: MxK bf16 row-major, W: NxK bf16 row-major)
// Block: 128 threads (4 waves), block tile 128x128, wave tile 64x64, K step 32.
// EPI: 0 = QKV scatter, 1 = +x -> fp32, 2 = +bias, gelu -> bf16, 3 = +bias +x -> fp32
template <int EPI>
__global__ __launch_bounds__(128) void gemm_wmma_kernel(
    const bf16_t* __restrict__ A, const bf16_t* __restrict__ W,
    int N, int K,
    const float* __restrict__ addsrc, const float* __restrict__ bias,
    float* __restrict__ outF, bf16_t* __restrict__ outB,
    bf16_t* __restrict__ qOut, bf16_t* __restrict__ kOut, bf16_t* __restrict__ vOut) {
  __shared__ __align__(16) bf16_t sA[2][128 * PAD_];
  __shared__ __align__(16) bf16_t sB[2][128 * PAD_];

  int tid  = threadIdx.x;
  int lane = tid & 31;
  int wv   = tid >> 5;
  int bM = blockIdx.y * 128, bN = blockIdx.x * 128;
  int wM = (wv >> 1) * 64, wN = (wv & 1) * 64;
  int ar = lane & 15;

  // each thread owns one row of the A tile and one row of the B tile
  const bf16_t* gArow = A + (size_t)(bM + tid) * K;
  const bf16_t* gBrow = W + (size_t)(bN + tid) * K;
  bf16_t* lArow[2] = { &sA[0][tid * PAD_], &sA[1][tid * PAD_] };
  bf16_t* lBrow[2] = { &sB[0][tid * PAD_], &sB[1][tid * PAD_] };

  auto stage = [&](int buf, int k0) {
    const bf16_t* ga = gArow + k0;
    const bf16_t* gb = gBrow + k0;
#pragma unroll
    for (int i = 0; i < 4; ++i) ASYNC_CP16(ga + i * 8, lArow[buf] + i * 8);
#pragma unroll
    for (int i = 0; i < 4; ++i) ASYNC_CP16(gb + i * 8, lBrow[buf] + i * 8);
  };

  f32x8 zero = {0.f, 0.f, 0.f, 0.f, 0.f, 0.f, 0.f, 0.f};
  f32x8 acc[4][4];
#pragma unroll
  for (int i = 0; i < 4; ++i)
#pragma unroll
    for (int j = 0; j < 4; ++j) acc[i][j] = zero;

  stage(0, 0);
  int nIter = K >> 5;
  for (int it = 0; it < nIter; ++it) {
    int buf = it & 1;
    WAIT_ASYNC0();          // my async fills for `buf` have landed
    __syncthreads();        // everyone's have landed
    if (it + 1 < nIter) stage(buf ^ 1, (it + 1) * 32);

    Frag a[4], b[4];
#pragma unroll
    for (int im = 0; im < 4; ++im)
      a[im] = load_a_frag(&sA[buf][(wM + 16 * im + ar) * PAD_], lane);
#pragma unroll
    for (int in = 0; in < 4; ++in)
      b[in] = load_b_frag(&sB[buf][(wN + 16 * in + ar) * PAD_], lane);
#pragma unroll
    for (int im = 0; im < 4; ++im)
#pragma unroll
      for (int in = 0; in < 4; ++in)
        acc[im][in] = wmma_bf16(a[im], b[in], acc[im][in]);
    __syncthreads();        // all waves done reading `buf` before it is refilled
  }

  int tM = bM + wM, tN = bN + wN;
  int half = lane >> 4;
  int cn   = lane & 15;
#pragma unroll
  for (int im = 0; im < 4; ++im)
#pragma unroll
    for (int in = 0; in < 4; ++in)
#pragma unroll
      for (int r = 0; r < 8; ++r) {
        float val = acc[im][in][r];
        int m = tM + 16 * im + half * 8 + r;
        int n = tN + 16 * in + cn;
        if constexpr (EPI == 0) {
          int sect = n >> 10;
          int hh   = (n & 1023) >> 6;
          int hd   = n & 63;
          int bb   = m >> 10;
          int l    = m & 1023;
          bf16_t bv = (bf16_t)val;
          if (sect == 0)      qOut[((size_t)(bb * H_ + hh) * L_ + l) * HD_ + hd] = bv;
          else if (sect == 1) kOut[((size_t)(bb * H_ + hh) * L_ + l) * HD_ + hd] = bv;
          else                vOut[((size_t)(bb * H_ + hh) * HD_ + hd) * L_ + l] = bv;
        } else if constexpr (EPI == 1) {
          outF[(size_t)m * N + n] = addsrc[(size_t)m * N + n] + val;
        } else if constexpr (EPI == 2) {
          float t = val + bias[n];
          outB[(size_t)m * N + n] = (bf16_t)gelu_exact(t);
        } else {
          outF[(size_t)m * N + n] = addsrc[(size_t)m * N + n] + val + bias[n];
        }
      }
}

// ---------------------------------------------------------------------------
// Flash attention: one wave per (b, h, 16-query tile).
// S^T = K.Q^T tiles -> softmax reduction is intra-lane + one shfl_xor(16);
// O^T = V^T.P^T with P^T rebuilt from accumulators via one cross-half shuffle.
__global__ __launch_bounds__(256) void attn_kernel(
    const bf16_t* __restrict__ qb, const bf16_t* __restrict__ kb,
    const bf16_t* __restrict__ vtb, const float* __restrict__ td,
    const float* __restrict__ tscale, const float* __restrict__ mbias,
    bf16_t* __restrict__ out) {
  int lane = threadIdx.x & 31;
  int wv   = threadIdx.x >> 5;
  int tile = blockIdx.x * 8 + wv;
  int qt = tile & 63;
  int h  = (tile >> 6) & 15;
  int b  = tile >> 10;
  int cn   = lane & 15;
  int half = lane >> 4;
  int q = qt * 16 + cn;

  const bf16_t* Q  = qb  + (size_t)(b * H_ + h) * L_ * HD_;
  const bf16_t* Kp = kb  + (size_t)(b * H_ + h) * L_ * HD_;
  const bf16_t* Vt = vtb + (size_t)(b * H_ + h) * HD_ * L_;

  Frag bq0 = load_b_frag(Q + (size_t)q * HD_ + 0, lane);
  Frag bq1 = load_b_frag(Q + (size_t)q * HD_ + 32, lane);

  f32x8 zero = {0.f, 0.f, 0.f, 0.f, 0.f, 0.f, 0.f, 0.f};
  f32x8 o[4];
#pragma unroll
  for (int t = 0; t < 4; ++t) o[t] = zero;

  float runM = -3.0e38f, runS = 0.f;
  const float sc = 0.125f;              // 1/sqrt(HD)
  float ts = tscale[h];
  const float* tdq = td + (size_t)q * L_;
  const float* mrow = mbias + (size_t)b * L_;

  for (int kc = 0; kc < L_; kc += 32) {
    // prefetch next chunk of K rows and V^T columns into the near (WGP) cache
    __builtin_prefetch(Kp + (size_t)(kc + 32 + cn) * HD_, 0, 3);
    __builtin_prefetch(Vt + (size_t)(cn) * L_ + kc + 32, 0, 3);

    f32x8 s0 = zero, s1 = zero;
    {
      Frag a0 = load_a_frag(Kp + (size_t)(kc + cn) * HD_ + 0, lane);
      s0 = wmma_bf16(a0, bq0, s0);
      Frag a1 = load_a_frag(Kp + (size_t)(kc + cn) * HD_ + 32, lane);
      s0 = wmma_bf16(a1, bq1, s0);
      Frag a2 = load_a_frag(Kp + (size_t)(kc + 16 + cn) * HD_ + 0, lane);
      s1 = wmma_bf16(a2, bq0, s1);
      Frag a3 = load_a_frag(Kp + (size_t)(kc + 16 + cn) * HD_ + 32, lane);
      s1 = wmma_bf16(a3, bq1, s1);
    }
    float p0[8], p1[8];
    int k0row = kc + half * 8;
    int k1row = kc + 16 + half * 8;
    float cmax = -3.0e38f;
#pragma unroll
    for (int r = 0; r < 8; ++r) {
      int ka = k0row + r, kb2 = k1row + r;
      float v0 = s0[r] * sc - tdq[ka] * ts + mrow[ka];
      float v1 = s1[r] * sc - tdq[kb2] * ts + mrow[kb2];
      p0[r] = v0; p1[r] = v1;
      cmax = fmaxf(cmax, fmaxf(v0, v1));
    }
    cmax = fmaxf(cmax, __shfl_xor(cmax, 16));
    float newM  = fmaxf(runM, cmax);
    float alpha = __expf(runM - newM);
    float csum = 0.f;
#pragma unroll
    for (int r = 0; r < 8; ++r) {
      p0[r] = (p0[r] < -1.0e37f) ? 0.f : __expf(p0[r] - newM);
      p1[r] = (p1[r] < -1.0e37f) ? 0.f : __expf(p1[r] - newM);
      csum += p0[r] + p1[r];
    }
    csum += __shfl_xor(csum, 16);
    runS = runS * alpha + csum;
    runM = newM;
#pragma unroll
    for (int t = 0; t < 4; ++t)
#pragma unroll
      for (int r = 0; r < 8; ++r) o[t][r] *= alpha;

    Frag bp;
#pragma unroll
    for (int r = 0; r < 8; ++r) {
      float po0 = __shfl_xor(p0[r], 16);
      float po1 = __shfl_xor(p1[r], 16);
      float lo = half ? po1 : p0[r];
      float hi = half ? p1[r] : po0;
      bp.h[r]     = (bf16_t)lo;
      bp.h[8 + r] = (bf16_t)hi;
    }
#pragma unroll
    for (int t = 0; t < 4; ++t) {
      Frag av = load_a_frag(Vt + (size_t)(t * 16 + cn) * L_ + kc, lane);
      o[t] = wmma_bf16(av, bp, o[t]);
    }
  }

  float inv = (runS > 0.f) ? (1.0f / runS) : 0.f;   // fully-masked -> 0 (nan_to_num)
#pragma unroll
  for (int t = 0; t < 4; ++t) {
    union { uint4 u; bf16_t hh[8]; } st;
#pragma unroll
    for (int r = 0; r < 8; ++r) st.hh[r] = (bf16_t)(o[t][r] * inv);
    *reinterpret_cast<uint4*>(out + ((size_t)(b * L_ + q) * D_) + h * HD_ + t * 16 + half * 8) = st.u;
  }
}

// ---------------------------------------------------------------------------
extern "C" void kernel_launch(void* const* d_in, const int* in_sizes, int n_in,
                              void* d_out, int out_size, void* d_ws, size_t ws_size,
                              hipStream_t stream) {
  const float* x        = (const float*)d_in[0];
  const float* torus    = (const float*)d_in[1];
  const float* time_emb = (const float*)d_in[2];
  const float* ln1_g    = (const float*)d_in[3];
  const float* ln1_b    = (const float*)d_in[4];
  const float* w_qkv    = (const float*)d_in[5];
  const float* w_out    = (const float*)d_in[6];
  const float* tscale   = (const float*)d_in[7];
  const float* ln2_g    = (const float*)d_in[8];
  const float* ln2_b    = (const float*)d_in[9];
  const float* w1       = (const float*)d_in[10];
  const float* b1       = (const float*)d_in[11];
  const float* w2       = (const float*)d_in[12];
  const float* b2       = (const float*)d_in[13];
  const float* w_time   = (const float*)d_in[14];
  const float* b_time   = (const float*)d_in[15];
  const unsigned char* mask = (const unsigned char*)d_in[16];
  float* outp = (float*)d_out;

  char* wsp = (char*)d_ws;
  auto alloc = [&](size_t bytes) -> void* {
    void* p = (void*)wsp;
    wsp += (bytes + 255) & ~(size_t)255;
    return p;
  };
  bf16_t* wqkv_bf = (bf16_t*)alloc((size_t)3 * D_ * D_ * 2);
  bf16_t* wout_bf = (bf16_t*)alloc((size_t)D_ * D_ * 2);
  bf16_t* w1_bf   = (bf16_t*)alloc((size_t)DFF_ * D_ * 2);
  bf16_t* w2_bf   = (bf16_t*)alloc((size_t)D_ * DFF_ * 2);
  float*  gelu_t  = (float*)alloc((size_t)B_ * D_ * 4);
  float*  tp      = (float*)alloc((size_t)B_ * 2 * D_ * 4);
  float*  mbias   = (float*)alloc((size_t)B_ * L_ * 4);
  bf16_t* h1_bf   = (bf16_t*)alloc((size_t)M_ * D_ * 2);
  bf16_t* q_bf    = (bf16_t*)alloc((size_t)M_ * D_ * 2);
  bf16_t* k_bf    = (bf16_t*)alloc((size_t)M_ * D_ * 2);
  bf16_t* vT_bf   = (bf16_t*)alloc((size_t)M_ * D_ * 2);
  bf16_t* attn_bf = (bf16_t*)alloc((size_t)M_ * D_ * 2);
  float*  x2      = (float*)alloc((size_t)M_ * D_ * 4);
  bf16_t* h2_bf   = (bf16_t*)alloc((size_t)M_ * D_ * 2);
  bf16_t* ff1_bf  = (bf16_t*)alloc((size_t)M_ * DFF_ * 2);

  // 1) weight conversions fp32 -> bf16; mask -> float bias
  {
    int n;
    n = 3 * D_ * D_; f2bf_kernel<<<(n + 255) / 256, 256, 0, stream>>>(w_qkv, wqkv_bf, n);
    n = D_ * D_;     f2bf_kernel<<<(n + 255) / 256, 256, 0, stream>>>(w_out, wout_bf, n);
    n = DFF_ * D_;   f2bf_kernel<<<(n + 255) / 256, 256, 0, stream>>>(w1, w1_bf, n);
    n = D_ * DFF_;   f2bf_kernel<<<(n + 255) / 256, 256, 0, stream>>>(w2, w2_bf, n);
    n = B_ * L_;     maskbias_kernel<<<(n + 255) / 256, 256, 0, stream>>>(mask, mbias, n);
  }
  // 2) time path
  gelu_vec_kernel<<<(B_ * D_ + 255) / 256, 256, 0, stream>>>(time_emb, gelu_t, B_ * D_);
  time_proj_kernel<<<(B_ * 2 * D_) / 256, 256, 0, stream>>>(gelu_t, w_time, b_time, tp);
  // 3) LN1 + adaLN modulation -> h1 (bf16)
  ln_mod_kernel<<<M_ / 8, 256, 0, stream>>>(x, ln1_g, ln1_b, tp, h1_bf, 1);
  // 4) QKV GEMM (M=8192, N=3072, K=1024) with per-head scatter epilogue
  gemm_wmma_kernel<0><<<dim3(3 * D_ / 128, M_ / 128), 128, 0, stream>>>(
      h1_bf, wqkv_bf, 3 * D_, D_, nullptr, nullptr, nullptr, nullptr, q_bf, k_bf, vT_bf);
  // 5) flash attention
  attn_kernel<<<(B_ * H_ * (L_ / 16)) / 8, 256, 0, stream>>>(
      q_bf, k_bf, vT_bf, torus, tscale, mbias, attn_bf);
  // 6) out projection + residual -> x2 (fp32)
  gemm_wmma_kernel<1><<<dim3(D_ / 128, M_ / 128), 128, 0, stream>>>(
      attn_bf, wout_bf, D_, D_, x, nullptr, x2, nullptr, nullptr, nullptr, nullptr);
  // 7) LN2 -> h2 (bf16)
  ln_mod_kernel<<<M_ / 8, 256, 0, stream>>>(x2, ln2_g, ln2_b, nullptr, h2_bf, 0);
  // 8) FFN1: gelu(h2 @ w1^T + b1) -> ff1 (bf16)
  gemm_wmma_kernel<2><<<dim3(DFF_ / 128, M_ / 128), 128, 0, stream>>>(
      h2_bf, w1_bf, DFF_, D_, nullptr, b1, nullptr, ff1_bf, nullptr, nullptr, nullptr);
  // 9) FFN2: out = x2 + ff1 @ w2^T + b2 (fp32)
  gemm_wmma_kernel<3><<<dim3(D_ / 128, M_ / 128), 128, 0, stream>>>(
      ff1_bf, w2_bf, D_, DFF_, x2, b2, outp, nullptr, nullptr, nullptr, nullptr);

  (void)in_sizes; (void)n_in; (void)out_size; (void)ws_size;
}
